// GraphNeuralConsensus_55825984913605
// MI455X (gfx1250) — matrix operand order repaired and checked by reading, not compiled, not run on hardware
//
#include <hip/hip_runtime.h>
#include <hip/hip_bf16.h>

#define N_AGENTS 8192
#define HID 64
#define NEG_BIG (-1.0e30f)

typedef __attribute__((ext_vector_type(16))) __bf16 v16bf;
typedef __attribute__((ext_vector_type(8)))  float  v8f;
typedef __attribute__((ext_vector_type(4)))  int    v4i;

static __device__ inline unsigned short f2bf_u16(float x) {
    union { __bf16 b; unsigned short u; } cv;
    cv.b = (__bf16)x;
    return cv.u;
}

// ---------------------------------------------------------------------------
// Kernel 1: per-layer dense transforms, specialized on DIN (4 or 64) so all
// tile loads are branch-free vectorized b128 loads.
// One wave per 16-row block:
//   h_self = h @ Ws^T + bs,  h_nb = h @ Wn^T + bn     (WMMA bf16, f32 accum)
//   s_row  = h_self @ a[:64] + ab,  s_col = h_nb @ a[64:]
// Stores h_self f32 [N][64], h_nb transposed bf16 [64][N], s_row/s_col f32.
// ---------------------------------------------------------------------------
template <int DIN>
__global__ __launch_bounds__(32)
void gnn_gemm_kernel(const float* __restrict__ hin,
                     const float* __restrict__ Ws, const float* __restrict__ bs,
                     const float* __restrict__ Wn, const float* __restrict__ bn,
                     const float* __restrict__ avec, const float* __restrict__ abp,
                     float* __restrict__ h_self, unsigned short* __restrict__ h_nbT,
                     float* __restrict__ s_row, float* __restrict__ s_col) {
    const int lane = threadIdx.x;          // 0..31
    const int l16  = lane & 15;
    const int half = lane >> 4;            // 0 or 1
    const int rb   = blockIdx.x * 16;
    const int row  = rb + l16;

    v8f cs[4], cn[4];
#pragma unroll
    for (int t = 0; t < 4; ++t) { cs[t] = (v8f)(0.0f); cn[t] = (v8f)(0.0f); }

#pragma unroll
    for (int kb = 0; kb < DIN; kb += 32) {
        // ---- A tile 16x32 (row = lane%16).  ISA layout:
        // lane<16 holds K 0-7 & 16-23, lane>=16 holds K 8-15 & 24-31.
        v16bf a;
        if constexpr (DIN >= 32) {
            const float* ab0 = hin + row * DIN + kb + half * 8;
            const float4 a0 = *(const float4*)(ab0);
            const float4 a1 = *(const float4*)(ab0 + 4);
            const float4 a2 = *(const float4*)(ab0 + 16);
            const float4 a3 = *(const float4*)(ab0 + 20);
            const float af[16] = {a0.x,a0.y,a0.z,a0.w, a1.x,a1.y,a1.z,a1.w,
                                  a2.x,a2.y,a2.z,a2.w, a3.x,a3.y,a3.z,a3.w};
#pragma unroll
            for (int e = 0; e < 16; ++e) a[e] = (__bf16)af[e];
        } else {
            // DIN==4: only lanes<16, elements 0..3 (K=0..3) are live.
            float4 av = make_float4(0.f, 0.f, 0.f, 0.f);
            if (half == 0) av = *(const float4*)(hin + row * DIN);
            a[0] = (__bf16)av.x; a[1] = (__bf16)av.y;
            a[2] = (__bf16)av.z; a[3] = (__bf16)av.w;
#pragma unroll
            for (int e = 4; e < 16; ++e) a[e] = (__bf16)0.0f;
        }
#pragma unroll
        for (int t = 0; t < 4; ++t) {
            const int col = t * 16 + l16;
            // ---- B tile 32x16: lane n = col n; lane<16 K 0-7 & 16-23, lane>=16 K 8-15 & 24-31.
            v16bf bS, bN;
            if constexpr (DIN >= 32) {
                const float* ws0 = Ws + col * DIN + kb + half * 8;
                const float* wn0 = Wn + col * DIN + kb + half * 8;
#pragma unroll
                for (int q = 0; q < 4; ++q) {
                    const float4 fs = *(const float4*)(ws0 + (q & 1) * 4 + (q >> 1) * 16);
                    const float4 fn = *(const float4*)(wn0 + (q & 1) * 4 + (q >> 1) * 16);
                    bS[q*4+0] = (__bf16)fs.x; bS[q*4+1] = (__bf16)fs.y;
                    bS[q*4+2] = (__bf16)fs.z; bS[q*4+3] = (__bf16)fs.w;
                    bN[q*4+0] = (__bf16)fn.x; bN[q*4+1] = (__bf16)fn.y;
                    bN[q*4+2] = (__bf16)fn.z; bN[q*4+3] = (__bf16)fn.w;
                }
            } else {
                // DIN==4: lanes<16 hold K=0..3 in elements 0..3, rest zero.
                float4 fs = make_float4(0.f, 0.f, 0.f, 0.f);
                float4 fn = make_float4(0.f, 0.f, 0.f, 0.f);
                if (half == 0) {
                    fs = *(const float4*)(Ws + col * DIN);
                    fn = *(const float4*)(Wn + col * DIN);
                }
                bS[0] = (__bf16)fs.x; bS[1] = (__bf16)fs.y;
                bS[2] = (__bf16)fs.z; bS[3] = (__bf16)fs.w;
                bN[0] = (__bf16)fn.x; bN[1] = (__bf16)fn.y;
                bN[2] = (__bf16)fn.z; bN[3] = (__bf16)fn.w;
#pragma unroll
                for (int e = 4; e < 16; ++e) { bS[e] = (__bf16)0.0f; bN[e] = (__bf16)0.0f; }
            }
            cs[t] = __builtin_amdgcn_wmma_f32_16x16x32_bf16(false, a, false, bS,
                                                            (short)0, cs[t], false, false);
            cn[t] = __builtin_amdgcn_wmma_f32_16x16x32_bf16(false, a, false, bN,
                                                            (short)0, cn[t], false, false);
        }
    }

    // bias add + attention-vector partial dots
    float srp[8], scp[8];
#pragma unroll
    for (int r = 0; r < 8; ++r) { srp[r] = 0.0f; scp[r] = 0.0f; }
#pragma unroll
    for (int t = 0; t < 4; ++t) {
        const int col = t * 16 + l16;
        const float bsv = bs[col], bnv = bn[col];
        const float a1 = avec[col], a2 = avec[HID + col];
#pragma unroll
        for (int r = 0; r < 8; ++r) {
            cs[t][r] += bsv;
            cn[t][r] += bnv;
            srp[r] += cs[t][r] * a1;
            scp[r] += cn[t][r] * a2;
        }
    }
    // reduce over the 16 columns held within each lane-half
    const float ab = abp[0];
#pragma unroll
    for (int r = 0; r < 8; ++r) {
#pragma unroll
        for (int off = 1; off < 16; off <<= 1) {
            srp[r] += __shfl_xor(srp[r], off, 32);
            scp[r] += __shfl_xor(scp[r], off, 32);
        }
        if (l16 == r) {
            const int orow = rb + r + half * 8;   // C-layout row for VGPR r
            s_row[orow] = srp[r] + ab;
            s_col[orow] = scp[r];
        }
    }

    // store h_self (f32, row-major) and h_nb (bf16, transposed [64][N])
#pragma unroll
    for (int t = 0; t < 4; ++t) {
        const int col = t * 16 + l16;
#pragma unroll
        for (int r = 0; r < 8; ++r) {
            const int orow = rb + r + half * 8;
            h_self[orow * HID + col] = cs[t][r];
            h_nbT[(size_t)col * N_AGENTS + orow] = f2bf_u16(cn[t][r]);
        }
    }
}

// ---------------------------------------------------------------------------
// Kernel 2: flash-attention style masked softmax-aggregate.
// 512 blocks x 128 threads (4 waves). Block owns 16 rows; waves split the
// 8192 columns in 32-wide tiles. adj is streamed with non-temporal loads
// (256 MB/layer, zero LRU reuse in 192 MB L2 -> keep L2 for h_nbT/s_col).
// ---------------------------------------------------------------------------
__global__ __launch_bounds__(128)
void gnn_attn_kernel(const int* __restrict__ adj,
                     const float* __restrict__ s_row, const float* __restrict__ s_col,
                     const unsigned short* __restrict__ h_nbT,
                     const float* __restrict__ h_self,
                     float* __restrict__ hout) {
    __shared__ float sm_m[4][16];
    __shared__ float sm_l[4][16];
    __shared__ float sm_acc[4][16][HID];

    const int tid  = threadIdx.x;
    const int wv   = tid >> 5;
    const int lane = tid & 31;
    const int l16  = lane & 15;
    const int half = lane >> 4;
    const int rb   = blockIdx.x * 16;
    const int row  = rb + l16;

    const float srow = s_row[row];
    const int* adjRow = adj + (size_t)row * N_AGENTS;

    float m = NEG_BIG;
    float l = 0.0f;
    v8f acc[4];
#pragma unroll
    for (int t = 0; t < 4; ++t) acc[t] = (v8f)(0.0f);

    for (int jt = wv * 32; jt < N_AGENTS; jt += 128) {
        const int J0 = jt + half * 8;   // this lane's two 8-wide K runs: J0, J0+16
        const v4i q0 = __builtin_nontemporal_load((const v4i*)(adjRow + J0));
        const v4i q1 = __builtin_nontemporal_load((const v4i*)(adjRow + J0 + 4));
        const v4i q2 = __builtin_nontemporal_load((const v4i*)(adjRow + J0 + 16));
        const v4i q3 = __builtin_nontemporal_load((const v4i*)(adjRow + J0 + 20));
        const float4 c0 = *(const float4*)(s_col + J0);
        const float4 c1 = *(const float4*)(s_col + J0 + 4);
        const float4 c2 = *(const float4*)(s_col + J0 + 16);
        const float4 c3 = *(const float4*)(s_col + J0 + 20);
        __builtin_prefetch(adjRow + J0 + 128, 0, 0);   // global_prefetch_b8: next tile

        float sc[16];
        const int   ai[16] = {q0.x,q0.y,q0.z,q0.w, q1.x,q1.y,q1.z,q1.w,
                              q2.x,q2.y,q2.z,q2.w, q3.x,q3.y,q3.z,q3.w};
        const float cv[16] = {c0.x,c0.y,c0.z,c0.w, c1.x,c1.y,c1.z,c1.w,
                              c2.x,c2.y,c2.z,c2.w, c3.x,c3.y,c3.z,c3.w};
        float bmax = NEG_BIG;
#pragma unroll
        for (int e = 0; e < 16; ++e) {
            sc[e] = (ai[e] > 0) ? (srow + cv[e]) : NEG_BIG;
            bmax = fmaxf(bmax, sc[e]);
        }
        bmax = fmaxf(bmax, __shfl_xor(bmax, 16, 32));   // combine K halves of row
        const float mnew = fmaxf(m, bmax);
        const float f = __expf(m - mnew);

        float psum = 0.0f;
        float p[16];
#pragma unroll
        for (int e = 0; e < 16; ++e) { p[e] = __expf(sc[e] - mnew); psum += p[e]; }
        psum += __shfl_xor(psum, 16, 32);
        l = l * f + psum;
        m = mnew;

        // rescale accumulators: VGPR r holds rows r (lanes<16) / r+8 (lanes>=16)
#pragma unroll
        for (int r = 0; r < 8; ++r) {
            const float fr = __shfl(f, r + half * 8, 32);
#pragma unroll
            for (int t = 0; t < 4; ++t) acc[t][r] *= fr;
        }

        // P tile already in A layout (element e -> K=(e&7)+8*half+16*(e>=8))
        v16bf pa;
#pragma unroll
        for (int e = 0; e < 16; ++e) pa[e] = (__bf16)p[e];

#pragma unroll
        for (int t = 0; t < 4; ++t) {
            const int col = t * 16 + l16;
            const v4i* bp = (const v4i*)(h_nbT + (size_t)col * N_AGENTS + jt + half * 8);
            union { v4i q[2]; v16bf v; } bu;
            bu.q[0] = bp[0];
            bu.q[1] = bp[2];
            acc[t] = __builtin_amdgcn_wmma_f32_16x16x32_bf16(false, pa, false, bu.v,
                                                             (short)0, acc[t], false, false);
        }
    }

    // stash per-wave partial softmax state
    sm_m[wv][l16] = m;
    sm_l[wv][l16] = l;
#pragma unroll
    for (int t = 0; t < 4; ++t)
#pragma unroll
        for (int r = 0; r < 8; ++r)
            sm_acc[wv][r + half * 8][t * 16 + l16] = acc[t][r];
    __syncthreads();

    // merge 4 waves, normalize, residual + ReLU
    for (int o = tid; o < 16 * HID; o += 128) {
        const int i = o >> 6;
        const int c = o & (HID - 1);
        float mm = sm_m[0][i];
#pragma unroll
        for (int w = 1; w < 4; ++w) mm = fmaxf(mm, sm_m[w][i]);
        float ll = 0.0f, ag = 0.0f;
#pragma unroll
        for (int w = 0; w < 4; ++w) {
            const float e = __expf(sm_m[w][i] - mm);
            ll += sm_l[w][i] * e;
            ag += sm_acc[w][i][c] * e;
        }
        const bool any_nb = (mm > -1.0e29f) && (ll > 0.0f);
        const float agg = any_nb ? (ag / ll) : 0.0f;
        const float v = h_self[(rb + i) * HID + c] + agg;
        hout[(rb + i) * HID + c] = fmaxf(v, 0.0f);
    }
}

// ---------------------------------------------------------------------------
// Kernel 3: final projection out = h @ Wout^T + bout  ([8192 x 4])
// ---------------------------------------------------------------------------
__global__ __launch_bounds__(128)
void gnn_out_kernel(const float* __restrict__ h,
                    const float* __restrict__ Wout, const float* __restrict__ bout,
                    float* __restrict__ out) {
    const int i = blockIdx.x * blockDim.x + threadIdx.x;
    if (i >= N_AGENTS) return;
    float hr[HID];
#pragma unroll
    for (int c = 0; c < HID; ++c) hr[c] = h[i * HID + c];
#pragma unroll
    for (int o = 0; o < 4; ++o) {
        float acc = bout[o];
#pragma unroll
        for (int c = 0; c < HID; ++c) acc += hr[c] * Wout[o * HID + c];
        out[i * 4 + o] = acc;
    }
}

// ---------------------------------------------------------------------------
// Host launcher. Input order (flattened setup_inputs dict):
//   0: states [8192,4] f32       1: adj [8192,8192] i32
//   2+6L..7+6L: layer L {Ws, bs, Wn, bn, a, ab}
//   20: Wout [4,64]              21: bout [4]
// ---------------------------------------------------------------------------
extern "C" void kernel_launch(void* const* d_in, const int* in_sizes, int n_in,
                              void* d_out, int out_size, void* d_ws, size_t ws_size,
                              hipStream_t stream) {
    (void)in_sizes; (void)n_in; (void)out_size; (void)ws_size;
    const float* states = (const float*)d_in[0];
    const int*   adj    = (const int*)d_in[1];

    char* ws = (char*)d_ws;
    float*          hbuf  = (float*)(ws);                           // 2 MB
    float*          hself = (float*)(ws + (2u << 20));              // 2 MB
    unsigned short* hnbT  = (unsigned short*)(ws + (4u << 20));     // 1 MB
    float*          srow  = (float*)(ws + (5u << 20));              // 32 KB
    float*          scol  = (float*)(ws + (5u << 20) + 32768);      // 32 KB

    for (int layer = 0; layer < 3; ++layer) {
        const int base = 2 + 6 * layer;
        const float* Ws = (const float*)d_in[base + 0];
        const float* bs = (const float*)d_in[base + 1];
        const float* Wn = (const float*)d_in[base + 2];
        const float* bn = (const float*)d_in[base + 3];
        const float* a  = (const float*)d_in[base + 4];
        const float* ab = (const float*)d_in[base + 5];

        if (layer == 0) {
            gnn_gemm_kernel<4><<<N_AGENTS / 16, 32, 0, stream>>>(
                states, Ws, bs, Wn, bn, a, ab, hself, hnbT, srow, scol);
        } else {
            gnn_gemm_kernel<HID><<<N_AGENTS / 16, 32, 0, stream>>>(
                hbuf, Ws, bs, Wn, bn, a, ab, hself, hnbT, srow, scol);
        }
        gnn_attn_kernel<<<N_AGENTS / 16, 128, 0, stream>>>(
            adj, srow, scol, hnbT, hself, hbuf);
    }

    const float* Wout = (const float*)d_in[20];
    const float* bout = (const float*)d_in[21];
    gnn_out_kernel<<<(N_AGENTS + 127) / 128, 128, 0, stream>>>(
        hbuf, Wout, bout, (float*)d_out);
}
